// GATEncoder_54288386621484
// MI455X (gfx1250) — matrix operand order repaired and checked by reading, not compiled, not run on hardware
//
#include <hip/hip_runtime.h>
#include <hip/hip_bf16.h>
#include <math.h>

typedef __bf16 bf16_t;
typedef __attribute__((ext_vector_type(16))) __bf16 v16bf;
typedef __attribute__((ext_vector_type(8)))  __bf16 v8bf;
typedef __attribute__((ext_vector_type(8)))  float  v8f;
typedef __attribute__((ext_vector_type(4)))  unsigned int u32x4;
typedef __attribute__((ext_vector_type(8)))  int i32x8;
typedef __attribute__((ext_vector_type(4)))  int i32x4;

#define DFEAT 768
#define NHEAD 8
#define CHEAD 96
#define NEG_SLOPE 0.2f

// ------------------------------------------------------------------
// GEMM: C[M,768] (f32) = A[M,768] (bf16, row-major) x B (bf16, COLUMN-major
// i.e. Bt[n*768+k]).  Tile 128x128, K-step 32, double-buffered LDS filled by
// the Tensor Data Mover (one tensor_load_to_lds per tile, wave 0 issues,
// TENSORcnt + one barrier per K-step).  8 waves; wave w computes the 16-row
// strip [w*16, w*16+16) x 128 cols as 8 WMMA 16x16 tiles.
// ------------------------------------------------------------------
#define TM 128
#define TN 128
#define TK 32
#define NT 8       // WMMA column tiles per wave (TN/16)
#define LDS_S 40   // bf16 stride per tile line: 32 data + 4-DWORD TDM pad

static __device__ __forceinline__ v16bf cat16(v8bf lo, v8bf hi) {
  return __builtin_shufflevector(lo, hi, 0,1,2,3,4,5,6,7,8,9,10,11,12,13,14,15);
}

#if __has_builtin(__builtin_amdgcn_tensor_load_to_lds)
#define USE_TDM 1
#else
#define USE_TDM 0
#endif

#if USE_TDM
// Loop-invariant descriptor group 1: data_size=2B, LDS pad 4 DWORDs every
// 16 DWORDs (-> LDS line stride 40 bf16), tile TK x t1, line stride DFEAT,
// tensor_dim0=DFEAT (k tiles always fit: no dim0 OOB possible),
// tensor_dim1=dim1 so the hardware zero-fills rows past the M edge.
static __device__ __forceinline__ i32x8 tdm_desc_g1(int dim1, int t1) {
  i32x8 g1;
  g1[0] = (1 << 16) | (1 << 20) | (3 << 22) | (3 << 25);
  g1[1] = (DFEAT & 0xFFFF) << 16;                   // tensor_dim0 lo
  g1[2] = (dim1 & 0xFFFF) << 16;                    // dim0 hi=0 | dim1 lo
  g1[3] = ((dim1 >> 16) & 0xFFFF) | (TK << 16);     // dim1 hi | tile_dim0
  g1[4] = t1 & 0xFFFF;                              // tile_dim1
  g1[5] = DFEAT;                                    // tensor_dim0_stride
  g1[6] = 0;
  g1[7] = 0;
  return g1;
}

// Per-iteration part: LDS destination + tile global address.
static __device__ __forceinline__ void tdm_tile_load(
    unsigned lds_addr, unsigned long long gaddr, i32x8 g1) {
  u32x4 g0;
  i32x4 gz4 = {0, 0, 0, 0};
  i32x8 gz8 = {0, 0, 0, 0, 0, 0, 0, 0};
  g0[0] = 1u;                                   // count=1, gather off
  g0[1] = lds_addr;                             // LDS byte address
  g0[2] = (unsigned)(gaddr & 0xFFFFFFFFull);    // global_addr[31:0]
  g0[3] = (unsigned)((gaddr >> 32) & 0x01FFFFFFull) | (2u << 30); // [56:32]|type=2
  __builtin_amdgcn_tensor_load_to_lds(g0, g1, gz4, gz4, gz8, 0);
}
#endif

__global__ __launch_bounds__(256) void gemm_bf16_wmma(
    const bf16_t* __restrict__ A, const bf16_t* __restrict__ Bt,
    float* __restrict__ C, int M)
{
  __shared__ __align__(128) bf16_t sA[2 * TM * LDS_S];
  __shared__ __align__(128) bf16_t sB[2 * TN * LDS_S];
  const int tid  = threadIdx.x;
  const int wave = tid >> 5;
  const int lane = tid & 31;
  const int m0 = blockIdx.x * TM;
  const int n0 = blockIdx.y * TN;
  const int NITER = DFEAT / TK;

  v8f acc[NT] = {};

#if USE_TDM
  const unsigned aBase = (unsigned)(size_t)(&sA[0]);
  const unsigned bBase = (unsigned)(size_t)(&sB[0]);
  const unsigned aBuf = TM * LDS_S * 2;  // bytes per A buffer
  const unsigned bBuf = TN * LDS_S * 2;  // bytes per B buffer
  const i32x8 g1A = tdm_desc_g1(M - m0, TM);      // zero-fill past row M
  const i32x8 g1B = tdm_desc_g1(DFEAT - n0, TN);
  unsigned long long gA = (unsigned long long)(size_t)(A + (size_t)m0 * DFEAT);
  unsigned long long gB = (unsigned long long)(size_t)(Bt + (size_t)n0 * DFEAT);

  if (wave == 0) {                 // prologue: fill buffer 0
    tdm_tile_load(aBase, gA, g1A);
    tdm_tile_load(bBase, gB, g1B);
  }
  gA += TK * 2;                    // next-tile addresses (bytes)
  gB += TK * 2;

  for (int it = 0; it < NITER; ++it) {
    __builtin_amdgcn_s_wait_tensorcnt(0);
    __syncthreads();               // tiles for `it` ready; prev reads retired
    const int cur = it & 1;
    if (wave == 0 && it + 1 < NITER) {
      tdm_tile_load(aBase + (cur ^ 1) * aBuf, gA, g1A);
      tdm_tile_load(bBase + (cur ^ 1) * bBuf, gB, g1B);
    }
    gA += TK * 2;
    gB += TK * 2;
    const bf16_t* sAc = sA + cur * TM * LDS_S;
    const bf16_t* sBc = sB + cur * TN * LDS_S;

    // A fragment (ISA 16-bit 16x32 layout): lanes 0-15: K 0..7 then 16..23;
    // lanes 16-31: K 8..15 then 24..31.
    const bf16_t* ap = sAc + (wave * 16 + (lane & 15)) * LDS_S + ((lane >> 4) << 3);
    v16bf af = cat16(*(const v8bf*)ap, *(const v8bf*)(ap + 16));
    #pragma unroll
    for (int nt = 0; nt < NT; ++nt) {
      // B fragment: lanes 0-15 -> K 0..7 then 16..23 of column N=lane; lanes 16-31 -> K 8..15 then 24..31
      const bf16_t* bp = sBc + (nt * 16 + (lane & 15)) * LDS_S + ((lane >> 4) << 3);
      v16bf bfrag = cat16(*(const v8bf*)bp, *(const v8bf*)(bp + 16));
      acc[nt] = __builtin_amdgcn_wmma_f32_16x16x32_bf16(
          false, af, false, bfrag, (short)0, acc[nt], false, false);
    }
  }
#else
  // Fallback: manual synchronous tile copies (single buffer).
  const int arow = tid >> 1;
  const int acol = (tid & 1) << 4;
  for (int k0 = 0; k0 < DFEAT; k0 += TK) {
    {
      v8bf u0 = {}, u1 = {};
      int grow = m0 + arow;
      if (grow < M) {
        const v8bf* gp = (const v8bf*)(A + (size_t)grow * DFEAT + k0 + acol);
        u0 = gp[0]; u1 = gp[1];
      }
      *(v8bf*)(sA + arow * LDS_S + acol)     = u0;
      *(v8bf*)(sA + arow * LDS_S + acol + 8) = u1;
    }
    {
      const v8bf* gp = (const v8bf*)(Bt + (size_t)(n0 + arow) * DFEAT + k0 + acol);
      v8bf u0 = gp[0], u1 = gp[1];
      *(v8bf*)(sB + arow * LDS_S + acol)     = u0;
      *(v8bf*)(sB + arow * LDS_S + acol + 8) = u1;
    }
    __syncthreads();
    const bf16_t* ap = sA + (wave * 16 + (lane & 15)) * LDS_S + ((lane >> 4) << 3);
    v16bf af = cat16(*(const v8bf*)ap, *(const v8bf*)(ap + 16));
    #pragma unroll
    for (int nt = 0; nt < NT; ++nt) {
      const bf16_t* bp = sB + (nt * 16 + (lane & 15)) * LDS_S + ((lane >> 4) << 3);
      v16bf bfrag = cat16(*(const v8bf*)bp, *(const v8bf*)(bp + 16));
      acc[nt] = __builtin_amdgcn_wmma_f32_16x16x32_bf16(
          false, af, false, bfrag, (short)0, acc[nt], false, false);
    }
    __syncthreads();
  }
#endif

  // C/D layout: VGPR j -> M=j (lanes 0-15) / M=j+8 (lanes 16-31); N=lane%16
  #pragma unroll
  for (int nt = 0; nt < NT; ++nt) {
    int col = n0 + nt * 16 + (lane & 15);
    #pragma unroll
    for (int j = 0; j < 8; ++j) {
      int r = m0 + wave * 16 + ((lane >> 4) << 3) + j;
      if (r < M) C[(size_t)r * DFEAT + col] = acc[nt][j];
    }
  }
}

// ------------------------------------------------------------------
// Small utility kernels
// ------------------------------------------------------------------
__global__ void k_fill(float* __restrict__ p, float v, long n) {
  long i = blockIdx.x * (long)blockDim.x + threadIdx.x;
  if (i < n) p[i] = v;
}

__global__ void k_f32_to_bf16(const float* __restrict__ s, bf16_t* __restrict__ d, long n) {
  long i = blockIdx.x * (long)blockDim.x + threadIdx.x;
  if (i < n) d[i] = (bf16_t)s[i];
}

// 768x768 weight: f32 row-major -> bf16 column-major (LDS-tiled transpose)
__global__ __launch_bounds__(256) void k_w_to_bf16_t(const float* __restrict__ s,
                                                     bf16_t* __restrict__ d) {
  __shared__ float t[32][33];
  int bx = blockIdx.x * 32, by = blockIdx.y * 32;
  int tx = threadIdx.x & 31, ty = threadIdx.x >> 5;
  #pragma unroll
  for (int r = ty; r < 32; r += 8)
    t[r][tx] = s[(size_t)(by + r) * DFEAT + bx + tx];
  __syncthreads();
  #pragma unroll
  for (int r = ty; r < 32; r += 8)
    d[(size_t)(bx + r) * DFEAT + by + tx] = (bf16_t)t[tx][r];
}

__global__ void k_bias_bf16(const float* __restrict__ s, const float* __restrict__ b,
                            bf16_t* __restrict__ d) {
  long i = (long)blockIdx.x * DFEAT + threadIdx.x;
  d[i] = (bf16_t)(s[i] + b[threadIdx.x]);
}

__global__ void k_relu_bias_bf16(const float* __restrict__ s, const float* __restrict__ b,
                                 bf16_t* __restrict__ d) {
  long i = (long)blockIdx.x * DFEAT + threadIdx.x;
  float v = s[i] + b[threadIdx.x];
  d[i] = (bf16_t)(v > 0.f ? v : 0.f);
}

__global__ void k_bias_out(const float* __restrict__ s, const float* __restrict__ b,
                           float* __restrict__ o) {
  long i = (long)blockIdx.x * DFEAT + threadIdx.x;
  o[i] = s[i] + b[threadIdx.x];
}

// alpha_{s,d}[n,h] = sum_c h[n,h,c] * a_{s,d}[h,c]  (one wave per head)
__global__ __launch_bounds__(256) void k_alpha(
    const float* __restrict__ h, const float* __restrict__ a_s,
    const float* __restrict__ a_d, float* __restrict__ os, float* __restrict__ od) {
  int n = blockIdx.x, hh = threadIdx.x >> 5, lane = threadIdx.x & 31;
  const float* hp  = h  + (size_t)n * DFEAT + hh * CHEAD;
  const float* asp = a_s + hh * CHEAD;
  const float* adp = a_d + hh * CHEAD;
  float s1 = 0.f, s2 = 0.f;
  for (int c = lane; c < CHEAD; c += 32) { float v = hp[c]; s1 += v * asp[c]; s2 += v * adp[c]; }
  for (int off = 16; off > 0; off >>= 1) { s1 += __shfl_down(s1, off); s2 += __shfl_down(s2, off); }
  if (lane == 0) { os[(size_t)n * NHEAD + hh] = s1; od[(size_t)n * NHEAD + hh] = s2; }
}

// ------------------------------------------------------------------
// Edge kernels (E real edges + N self loops appended)
// ------------------------------------------------------------------
__global__ void k_deg(const long long* __restrict__ ei, int E, int Nn, float* __restrict__ deg) {
  long t = blockIdx.x * (long)blockDim.x + threadIdx.x;
  if (t >= (long)E + Nn) return;
  int d = (t < E) ? (int)ei[E + t] : (int)(t - E);
  unsafeAtomicAdd(&deg[d], 1.0f);
}

__global__ void k_dinv(const float* __restrict__ deg, float* __restrict__ dinv, int Nn) {
  int n = blockIdx.x * blockDim.x + threadIdx.x;
  if (n < Nn) dinv[n] = deg[n] > 0.f ? rsqrtf(deg[n]) : 0.f;
}

__global__ void k_edge_max(const long long* __restrict__ ei, int E, int Nn,
                           const float* __restrict__ as, const float* __restrict__ ad,
                           float* __restrict__ mx) {
  long t = blockIdx.x * (long)blockDim.x + threadIdx.x;
  if (t >= (long)(E + Nn) * NHEAD) return;
  long e = t >> 3; int hh = (int)(t & 7);
  int s, d;
  if (e < E) { s = (int)ei[e]; d = (int)ei[E + e]; } else { s = d = (int)(e - E); }
  float v = as[(size_t)s * NHEAD + hh] + ad[(size_t)d * NHEAD + hh];
  v = v > 0.f ? v : NEG_SLOPE * v;
  __hip_atomic_fetch_max(&mx[(size_t)d * NHEAD + hh], v,
                         __ATOMIC_RELAXED, __HIP_MEMORY_SCOPE_AGENT);
}

__global__ void k_edge_exp(const long long* __restrict__ ei, int E, int Nn,
                           const float* __restrict__ as, const float* __restrict__ ad,
                           const float* __restrict__ mx, float* __restrict__ ew,
                           float* __restrict__ den) {
  long t = blockIdx.x * (long)blockDim.x + threadIdx.x;
  if (t >= (long)(E + Nn) * NHEAD) return;
  long e = t >> 3; int hh = (int)(t & 7);
  int s, d;
  if (e < E) { s = (int)ei[e]; d = (int)ei[E + e]; } else { s = d = (int)(e - E); }
  float v = as[(size_t)s * NHEAD + hh] + ad[(size_t)d * NHEAD + hh];
  v = v > 0.f ? v : NEG_SLOPE * v;
  float wv = __expf(v - mx[(size_t)d * NHEAD + hh]);
  ew[(size_t)e * NHEAD + hh] = wv;
  unsafeAtomicAdd(&den[(size_t)d * NHEAD + hh], wv);
}

// one 768-thread block per edge: out[dst,:] += h[src,:] * alpha
__global__ void k_gat_scatter(const long long* __restrict__ ei, int E, int Nn,
                              const float* __restrict__ h, const float* __restrict__ ew,
                              const float* __restrict__ den, float* __restrict__ out) {
  int e = blockIdx.x, c = threadIdx.x;
  int s, d;
  if (e < E) { s = (int)ei[e]; d = (int)ei[E + e]; } else { s = d = e - E; }
  int hh = c / CHEAD;
  float coef = ew[(size_t)e * NHEAD + hh] / den[(size_t)d * NHEAD + hh];
  unsafeAtomicAdd(&out[(size_t)d * DFEAT + c], h[(size_t)s * DFEAT + c] * coef);
}

__global__ void k_gcn_scatter(const long long* __restrict__ ei, int E, int Nn,
                              const float* __restrict__ h, const float* __restrict__ dinv,
                              float* __restrict__ out) {
  int e = blockIdx.x, c = threadIdx.x;
  int s, d;
  if (e < E) { s = (int)ei[e]; d = (int)ei[E + e]; } else { s = d = e - E; }
  float coef = dinv[s] * dinv[d];
  unsafeAtomicAdd(&out[(size_t)d * DFEAT + c], h[(size_t)s * DFEAT + c] * coef);
}

// ------------------------------------------------------------------
extern "C" void kernel_launch(void* const* d_in, const int* in_sizes, int n_in,
                              void* d_out, int out_size, void* d_ws, size_t ws_size,
                              hipStream_t stream) {
  const float*     x    = (const float*)d_in[0];
  const long long* ei   = (const long long*)d_in[1];
  const float*     W0   = (const float*)d_in[2];
  const float*     a_s0 = (const float*)d_in[3];
  const float*     a_d0 = (const float*)d_in[4];
  const float*     b0   = (const float*)d_in[5];
  const float*     W1   = (const float*)d_in[6];
  const float*     a_s1 = (const float*)d_in[7];
  const float*     a_d1 = (const float*)d_in[8];
  const float*     b1   = (const float*)d_in[9];
  const float*     Ws   = (const float*)d_in[10];
  const float*     bs   = (const float*)d_in[11];
  const float*     Wmu  = (const float*)d_in[12];
  const float*     bmu  = (const float*)d_in[13];
  const float*     Wlv  = (const float*)d_in[14];
  const float*     blv  = (const float*)d_in[15];

  const int N  = in_sizes[0] / DFEAT;
  const int E  = in_sizes[1] / 2;
  const int EE = E + N;
  const long nx = (long)N * DFEAT;
  const long nh = (long)N * NHEAD;
  const long et = (long)EE * NHEAD;

  // workspace carve-out (256B-aligned slices)
  char* wp = (char*)d_ws;
  auto carve = [&](size_t bytes) -> char* {
    char* p = wp; wp += (bytes + 255) & ~(size_t)255; return p;
  };
  float*  hA   = (float*)carve((size_t)nx * 4);
  float*  hB   = (float*)carve((size_t)nx * 4);
  bf16_t* Xb   = (bf16_t*)carve((size_t)nx * 2);
  bf16_t* Wb   = (bf16_t*)carve((size_t)DFEAT * DFEAT * 2);
  float*  asb  = (float*)carve((size_t)nh * 4);
  float*  adb  = (float*)carve((size_t)nh * 4);
  float*  mmax = (float*)carve((size_t)nh * 4);
  float*  den  = (float*)carve((size_t)nh * 4);
  float*  ew   = (float*)carve((size_t)et * 4);
  float*  deg  = (float*)carve((size_t)N * 4);
  float*  dinv = (float*)carve((size_t)N * 4);

  dim3 gg((N + TM - 1) / TM, DFEAT / TN);
  dim3 wt(DFEAT / 32, DFEAT / 32);
  auto fill = [&](float* p, float v, long n) {
    k_fill<<<(n + 1023) / 1024, 1024, 0, stream>>>(p, v, n);
  };

  // degrees (with self loops) -> dinv
  fill(deg, 0.f, N);
  k_deg<<<(EE + 255) / 256, 256, 0, stream>>>(ei, E, N, deg);
  k_dinv<<<(N + 255) / 256, 256, 0, stream>>>(deg, dinv, N);

  // ---------------- GAT layer 0 ----------------
  k_f32_to_bf16<<<(nx + 255) / 256, 256, 0, stream>>>(x, Xb, nx);
  k_w_to_bf16_t<<<wt, 256, 0, stream>>>(W0, Wb);
  gemm_bf16_wmma<<<gg, 256, 0, stream>>>(Xb, Wb, hA, N);
  k_alpha<<<N, 256, 0, stream>>>(hA, a_s0, a_d0, asb, adb);
  fill(mmax, -3.0e38f, nh); fill(den, 0.f, nh); fill(hB, 0.f, nx);
  k_edge_max<<<(et + 255) / 256, 256, 0, stream>>>(ei, E, N, asb, adb, mmax);
  k_edge_exp<<<(et + 255) / 256, 256, 0, stream>>>(ei, E, N, asb, adb, mmax, ew, den);
  k_gat_scatter<<<EE, DFEAT, 0, stream>>>(ei, E, N, hA, ew, den, hB);
  k_bias_bf16<<<N, DFEAT, 0, stream>>>(hB, b0, Xb);

  // ---------------- GAT layer 1 ----------------
  k_w_to_bf16_t<<<wt, 256, 0, stream>>>(W1, Wb);
  gemm_bf16_wmma<<<gg, 256, 0, stream>>>(Xb, Wb, hA, N);
  k_alpha<<<N, 256, 0, stream>>>(hA, a_s1, a_d1, asb, adb);
  fill(mmax, -3.0e38f, nh); fill(den, 0.f, nh); fill(hB, 0.f, nx);
  k_edge_max<<<(et + 255) / 256, 256, 0, stream>>>(ei, E, N, asb, adb, mmax);
  k_edge_exp<<<(et + 255) / 256, 256, 0, stream>>>(ei, E, N, asb, adb, mmax, ew, den);
  k_gat_scatter<<<EE, DFEAT, 0, stream>>>(ei, E, N, hA, ew, den, hB);
  k_bias_bf16<<<N, DFEAT, 0, stream>>>(hB, b1, Xb);

  // ---------------- GCN shared (relu) ----------------
  k_w_to_bf16_t<<<wt, 256, 0, stream>>>(Ws, Wb);
  gemm_bf16_wmma<<<gg, 256, 0, stream>>>(Xb, Wb, hA, N);
  fill(hB, 0.f, nx);
  k_gcn_scatter<<<EE, DFEAT, 0, stream>>>(ei, E, N, hA, dinv, hB);
  k_relu_bias_bf16<<<N, DFEAT, 0, stream>>>(hB, bs, Xb);

  // ---------------- GCN mu ----------------
  k_w_to_bf16_t<<<wt, 256, 0, stream>>>(Wmu, Wb);
  gemm_bf16_wmma<<<gg, 256, 0, stream>>>(Xb, Wb, hA, N);
  fill(hB, 0.f, nx);
  k_gcn_scatter<<<EE, DFEAT, 0, stream>>>(ei, E, N, hA, dinv, hB);
  k_bias_out<<<N, DFEAT, 0, stream>>>(hB, bmu, (float*)d_out);

  // ---------------- GCN logvar ----------------
  k_w_to_bf16_t<<<wt, 256, 0, stream>>>(Wlv, Wb);
  gemm_bf16_wmma<<<gg, 256, 0, stream>>>(Xb, Wb, hA, N);
  fill(hB, 0.f, nx);
  k_gcn_scatter<<<EE, DFEAT, 0, stream>>>(ei, E, N, hA, dinv, hB);
  k_bias_out<<<N, DFEAT, 0, stream>>>(hB, blv, (float*)d_out + (size_t)N * DFEAT);
}
